// LuongAttention_12893491822631
// MI455X (gfx1250) — matrix operand order repaired and checked by reading, not compiled, not run on hardware
//
#include <hip/hip_runtime.h>
#include <hip/hip_bf16.h>
#include <stdint.h>

typedef __bf16 bf16;
typedef __attribute__((ext_vector_type(16))) __bf16 v16bf;
typedef __attribute__((ext_vector_type(8)))  __bf16 v8bf;
typedef __attribute__((ext_vector_type(4)))  __bf16 v4bf;
typedef __attribute__((ext_vector_type(8)))  float  v8f;

typedef __attribute__((ext_vector_type(4))) unsigned int tdm_g0_t;
typedef __attribute__((ext_vector_type(8))) int          tdm_g1_t;
typedef __attribute__((ext_vector_type(4))) int          tdm_g2_t;
typedef __attribute__((ext_vector_type(8))) int          tdm_g4_t;

static constexpr int BATCH = 32;
static constexpr int TQ    = 1024;
static constexpr int TK    = 1024;
static constexpr int HDIM  = 512;

static __device__ __forceinline__ v16bf cat8(v8bf a, v8bf b) {
  return __builtin_shufflevector(a, b, 0,1,2,3,4,5,6,7,8,9,10,11,12,13,14,15);
}
static __device__ __forceinline__ v8f wmma_bf16(v16bf a, v16bf b, v8f c) {
  // D = A(16x32 bf16) * B(32x16 bf16) + C(16x16 f32)
  return __builtin_amdgcn_wmma_f32_16x16x32_bf16(false, a, false, b, (short)0, c,
                                                 false, false);
}

// ---------------------------------------------------------------------------
// TDM: async 2D tile load (128 rows x 32 bf16, row stride = k_elems) into LDS.
// pad_enable inserts 4 DWORDs (16B) after every 16 DWORDs (64B = one row),
// giving an 80B LDS row stride (== bf16[40] padded rows, bank-conflict-free).
// D# layout per CDNA5 ISA ch.8 (group0: 128b, group1: 256b, groups2/3: zero).
// This toolchain exposes the 6-arg builtin (extra zero int32x8 group + cpol).
// ---------------------------------------------------------------------------
static __device__ __forceinline__ void tdm_load_b_tile(unsigned lds_byte_off,
                                                       const void* gaddr,
                                                       int k_elems) {
  unsigned long long ga = (unsigned long long)(uintptr_t)gaddr;
  tdm_g0_t g0 = { 1u,                                   // count=1 (valid D#)
                  lds_byte_off,                         // lds_addr
                  (unsigned)ga,                         // global_addr[31:0]
                  (unsigned)((ga >> 32) & 0x01ffffffu)  // global_addr[56:32]
                      | (2u << 30) };                   // type=2 ("image")
  tdm_g1_t g1 = {
      (int)((1u << 16)      // data_size = 2 bytes
            | (1u << 20)    // pad_enable
            | (3u << 22)    // pad_interval: 16 DWORDs (64B)
            | (3u << 25)),  // pad_amount:   4 DWORDs (16B)
      (int)(((unsigned)k_elems & 0xffffu) << 16),          // tensor_dim0 lo16
      (int)((((unsigned)k_elems >> 16) & 0xffffu)          // tensor_dim0 hi16
            | (128u << 16)),                               // tensor_dim1 lo16
      (int)(32u << 16),                 // tensor_dim1 hi=0 | tile_dim0=32
      (int)128,                         // tile_dim1=128 | tile_dim2=0
      (int)k_elems,                     // tensor_dim0_stride[31:0]
      0,                                // stride0 hi | stride1 lo
      0 };                              // stride1 hi
  tdm_g2_t gz4 = { 0, 0, 0, 0 };
  tdm_g4_t gz8 = { 0, 0, 0, 0, 0, 0, 0, 0 };
  __builtin_amdgcn_tensor_load_to_lds(g0, g1, gz4, gz4, gz8, 0);
}

// ---------------------------------------------------------------------------
// fp32 -> bf16 elementwise convert, 4 elems / thread (exact sizes, no tail)
// ---------------------------------------------------------------------------
__global__ void cvt_f32_bf16_x4(const float* __restrict__ src, bf16* __restrict__ dst) {
  size_t i = (size_t)blockIdx.x * blockDim.x + threadIdx.x;
  float4 v = ((const float4*)src)[i];
  v4bf o = { (bf16)v.x, (bf16)v.y, (bf16)v.z, (bf16)v.w };
  ((v4bf*)dst)[i] = o;
}

// ---------------------------------------------------------------------------
// keys: fp32 [b, j, h] -> bf16 [b, j, h]  AND  bf16 transposed [b, h, j]
// ---------------------------------------------------------------------------
__global__ void keys_cvt_transpose(const float* __restrict__ keys,
                                   bf16* __restrict__ keys_bf,
                                   bf16* __restrict__ keysT_bf) {
  __shared__ bf16 tile[32][33];
  const int b  = blockIdx.z;
  const int j0 = blockIdx.x * 32;
  const int h0 = blockIdx.y * 32;
  const int tx = threadIdx.x, ty = threadIdx.y;
  const float* kb  = keys     + (size_t)b * TK * HDIM;
  bf16*        kbf = keys_bf  + (size_t)b * TK * HDIM;
  bf16*        ktf = keysT_bf + (size_t)b * HDIM * TK;
#pragma unroll
  for (int i = 0; i < 4; ++i) {
    int j = j0 + ty + 8 * i;
    bf16 v = (bf16)kb[(size_t)j * HDIM + h0 + tx];
    tile[ty + 8 * i][tx] = v;
    kbf[(size_t)j * HDIM + h0 + tx] = v;
  }
  __syncthreads();
#pragma unroll
  for (int i = 0; i < 4; ++i) {
    int h = h0 + ty + 8 * i;
    ktf[(size_t)h * TK + j0 + tx] = tile[tx][ty + 8 * i];
  }
}

// ---------------------------------------------------------------------------
// q-projection: Q[m,n] = sum_k query_bf[m,k] * W_bf[n,k] + bias[n], bf16 out
// M = B*Tq = 32768, N = K = 512. One wave -> 32x64 tile (W is L2-resident).
// ---------------------------------------------------------------------------
__global__ void qproj_wmma(const bf16* __restrict__ A,   // [32768, 512]
                           const bf16* __restrict__ Wb,  // [512(n), 512(k)]
                           const float* __restrict__ bias,
                           bf16* __restrict__ Q) {       // [32768, 512] bf16
  const int K = HDIM, N = HDIM;
  const int w    = (blockIdx.x * blockDim.x + threadIdx.x) >> 5;
  const int lane = threadIdx.x & 31;
  const int l = lane & 15, hi = lane >> 4;
  const int m0 = (w >> 3) * 32;
  const int n0 = (w & 7) * 64;
  const bf16* arow0 = A + (size_t)(m0 + l) * K;
  const bf16* arow1 = arow0 + (size_t)16 * K;
  v8f acc[2][4] = {};
  for (int k0 = 0; k0 < K; k0 += 32) {
    __builtin_prefetch(arow0 + k0 + 256, 0, 3);
    __builtin_prefetch(arow1 + k0 + 256, 0, 3);
    v16bf a0 = cat8(*(const v8bf*)(arow0 + k0 + hi * 8),
                    *(const v8bf*)(arow0 + k0 + 16 + hi * 8));
    v16bf a1 = cat8(*(const v8bf*)(arow1 + k0 + hi * 8),
                    *(const v8bf*)(arow1 + k0 + 16 + hi * 8));
#pragma unroll
    for (int t = 0; t < 4; ++t) {
      const bf16* brow = Wb + (size_t)(n0 + t * 16 + l) * K + k0 + hi * 8;
      v16bf bfrag = cat8(*(const v8bf*)brow, *(const v8bf*)(brow + 16));
      acc[0][t] = wmma_bf16(a0, bfrag, acc[0][t]);
      acc[1][t] = wmma_bf16(a1, bfrag, acc[1][t]);
    }
  }
#pragma unroll
  for (int t = 0; t < 4; ++t) {
    int n = n0 + t * 16 + l;
    float bv = bias[n];
#pragma unroll
    for (int r = 0; r < 8; ++r) {
      Q[(size_t)(m0 + hi * 8 + r) * N + n]      = (bf16)(acc[0][t][r] + bv);
      Q[(size_t)(m0 + 16 + hi * 8 + r) * N + n] = (bf16)(acc[1][t][r] + bv);
    }
  }
}

// ---------------------------------------------------------------------------
// Batched tiled GEMM: C[b] = A[b] (TQ x K, bf16) * Bt[b]^T (Bt is N x K, bf16)
// fp32 out. Block = 8 waves -> 128x128 C tile; wave -> 32x64.
// B slab (32 x 128 per K-step, 8KB) DMA'd into LDS by the TDM (wave 0 issues
// one descriptor per K-step, double buffered, tracked with TENSORcnt).
// TDM padding gives 80B LDS rows -> conflict-free b128 fragment reads.
// ---------------------------------------------------------------------------
template <int K, int N>
__global__ void gemm128x128_wmma(const bf16* __restrict__ Aall,
                                 const bf16* __restrict__ Btall,
                                 float* __restrict__ Call) {
  constexpr int NB    = N / 128;   // column-blocks
  constexpr int NSTEP = K / 32;
  __shared__ bf16 ldsB[2][128][40];

  const int b  = blockIdx.y;
  const int mb = blockIdx.x / NB;
  const int nb = blockIdx.x % NB;
  const bf16* A  = Aall  + (size_t)b * TQ * K;
  const bf16* Bt = Btall + (size_t)b * N * K;
  float*      C  = Call  + (size_t)b * TQ * N;

  const int tid  = threadIdx.x;
  const int lane = tid & 31;
  const int wid  = tid >> 5;
  const int l = lane & 15, hi = lane >> 4;
  const int wm = wid & 3, wn = wid >> 2;         // 4 x 2 wave grid
  const int m0   = mb * 128 + wm * 32;
  const int nblk = nb * 128;

  // wave-relative LDS byte offsets of the two B buffers (flat addr low bits)
  const unsigned ldsOff[2] = { (unsigned)(uintptr_t)&ldsB[0][0][0],
                               (unsigned)(uintptr_t)&ldsB[1][0][0] };
  const bf16* btile = Bt + (size_t)nblk * K;     // tile origin: row nblk, k=0

  const bf16* arow0 = A + (size_t)(m0 + l) * K;
  const bf16* arow1 = arow0 + (size_t)16 * K;

  v8f acc[2][4] = {};

  // preload B tile 0 via TDM
  if (wid == 0) {
    tdm_load_b_tile(ldsOff[0], btile, K);
    __builtin_amdgcn_s_wait_tensorcnt(0);
  }
  __syncthreads();

  for (int i = 0; i < NSTEP; ++i) {
    const int k0  = i * 32;
    const int cur = i & 1;
    const int nxt = cur ^ 1;

    // kick off next B slab DMA (runs while we do the math below)
    if (i + 1 < NSTEP && wid == 0)
      tdm_load_b_tile(ldsOff[nxt], btile + (k0 + 32), K);

    __builtin_prefetch(arow0 + k0 + 256, 0, 3);
    v16bf a0 = cat8(*(const v8bf*)(arow0 + k0 + hi * 8),
                    *(const v8bf*)(arow0 + k0 + 16 + hi * 8));
    v16bf a1 = cat8(*(const v8bf*)(arow1 + k0 + hi * 8),
                    *(const v8bf*)(arow1 + k0 + 16 + hi * 8));

#pragma unroll
    for (int t = 0; t < 4; ++t) {
      const bf16* bp = &ldsB[cur][wn * 64 + t * 16 + l][hi * 8];
      v16bf bfrag = cat8(*(const v8bf*)bp, *(const v8bf*)(bp + 16));
      acc[0][t] = wmma_bf16(a0, bfrag, acc[0][t]);
      acc[1][t] = wmma_bf16(a1, bfrag, acc[1][t]);
    }

    if (i + 1 < NSTEP) {
      if (wid == 0) __builtin_amdgcn_s_wait_tensorcnt(0);
      __syncthreads();
    }
  }

#pragma unroll
  for (int t = 0; t < 4; ++t) {
    int n = nblk + wn * 64 + t * 16 + l;
#pragma unroll
    for (int r = 0; r < 8; ++r) {
      C[(size_t)(m0 + hi * 8 + r) * N + n]      = acc[0][t][r];
      C[(size_t)(m0 + 16 + hi * 8 + r) * N + n] = acc[1][t][r];
    }
  }
}

// ---------------------------------------------------------------------------
// row softmax over Tk=1024, in place (fp32) + bf16 copy for the next GEMM.
// ---------------------------------------------------------------------------
__global__ void softmax_row(float* __restrict__ S, bf16* __restrict__ Sbf) {
  const size_t row = blockIdx.x;
  float* p  = S   + row * TK;
  bf16*  pb = Sbf + row * TK;
  const int tid = threadIdx.x;
  float4 v = ((const float4*)p)[tid];
  float m = fmaxf(fmaxf(v.x, v.y), fmaxf(v.z, v.w));
#pragma unroll
  for (int off = 16; off > 0; off >>= 1) m = fmaxf(m, __shfl_xor(m, off, 32));
  __shared__ float red[8];
  if ((tid & 31) == 0) red[tid >> 5] = m;
  __syncthreads();
  float bm = red[0];
#pragma unroll
  for (int i = 1; i < 8; ++i) bm = fmaxf(bm, red[i]);
  __syncthreads();
  float e0 = __expf(v.x - bm), e1 = __expf(v.y - bm);
  float e2 = __expf(v.z - bm), e3 = __expf(v.w - bm);
  float s = e0 + e1 + e2 + e3;
#pragma unroll
  for (int off = 16; off > 0; off >>= 1) s += __shfl_xor(s, off, 32);
  if ((tid & 31) == 0) red[tid >> 5] = s;
  __syncthreads();
  float tot = 0.f;
#pragma unroll
  for (int i = 0; i < 8; ++i) tot += red[i];
  float inv = 1.0f / tot;
  float4 o = { e0 * inv, e1 * inv, e2 * inv, e3 * inv };
  ((float4*)p)[tid] = o;
  v4bf ob = { (bf16)o.x, (bf16)o.y, (bf16)o.z, (bf16)o.w };
  ((v4bf*)pb)[tid] = ob;
}

// ---------------------------------------------------------------------------
extern "C" void kernel_launch(void* const* d_in, const int* in_sizes, int n_in,
                              void* d_out, int out_size, void* d_ws, size_t ws_size,
                              hipStream_t stream) {
  (void)in_sizes; (void)n_in; (void)out_size; (void)ws_size;
  const float* query = (const float*)d_in[0];  // [32,1024,512]
  const float* keys  = (const float*)d_in[1];  // [32,1024,512]
  const float* W     = (const float*)d_in[2];  // [512,512] (out,in)
  const float* bias  = (const float*)d_in[3];  // [512]

  float* out_ctx  = (float*)d_out;                             // [32,1024,512]
  float* out_attn = out_ctx + (size_t)BATCH * TQ * HDIM;       // [32,1024,1024]

  char* ws = (char*)d_ws;
  size_t off = 0;
  bf16* query_bf = (bf16*)(ws + off); off += (size_t)BATCH * TQ * HDIM * 2;
  bf16* keys_bf  = (bf16*)(ws + off); off += (size_t)BATCH * TK * HDIM * 2;
  bf16* keysT_bf = (bf16*)(ws + off); off += (size_t)BATCH * HDIM * TK * 2;
  bf16* W_bf     = (bf16*)(ws + off); off += (size_t)HDIM * HDIM * 2;
  bf16* q_bf     = (bf16*)(ws + off); off += (size_t)BATCH * TQ * HDIM * 2;
  bf16* attn_bf  = (bf16*)(ws + off); off += (size_t)BATCH * TQ * TK * 2;

  // 1) fp32 -> bf16 converts
  cvt_f32_bf16_x4<<<(size_t)BATCH * TQ * HDIM / 4 / 256, 256, 0, stream>>>(query, query_bf);
  cvt_f32_bf16_x4<<<(size_t)HDIM * HDIM / 4 / 256, 256, 0, stream>>>(W, W_bf);

  // 2) keys convert + transpose
  keys_cvt_transpose<<<dim3(TK / 32, HDIM / 32, BATCH), dim3(32, 8), 0, stream>>>(
      keys, keys_bf, keysT_bf);

  // 3) q = query @ W^T + b   (32x64 waves, 8192 waves -> 1024 blocks)
  qproj_wmma<<<(BATCH * TQ / 32) * (HDIM / 64) / 8, 256, 0, stream>>>(
      query_bf, W_bf, bias, q_bf);

  // 4) scores = q @ keys^T   (128x128 block tiles: 64 blocks x 32 batches)
  gemm128x128_wmma<HDIM, TK>
      <<<dim3((TQ / 128) * (TK / 128), BATCH), 256, 0, stream>>>(
          q_bf, keys_bf, out_attn);

  // 5) softmax rows in place + bf16 copy (32768 rows)
  softmax_row<<<BATCH * TQ, 256, 0, stream>>>(out_attn, attn_bf);

  // 6) context = attn @ keys (128x128 block tiles: 32 blocks x 32 batches)
  gemm128x128_wmma<TK, HDIM>
      <<<dim3((TQ / 128) * (HDIM / 128), BATCH), 256, 0, stream>>>(
          attn_bf, keysT_bf, out_ctx);
}
